// QuantumQKGenerator_47777216201066
// MI455X (gfx1250) — matrix operand order repaired and checked by reading, not compiled, not run on hardware
//
#include <hip/hip_runtime.h>
#include <hip/hip_bf16.h>
#include <math.h>

typedef __attribute__((ext_vector_type(16))) _Float16 v16h;
typedef __attribute__((ext_vector_type(8)))  float    v8f;

#define NQ    10
#define DIM   1024
#define BATCH 8192
#define K2    2048      // 2*DIM (block-complex embedding)
#define NB_K  64        // K2/32 k-tiles
#define NB_N  128       // K2/16 n-tiles

// ---- WMMA operand swizzles (we control producer AND consumer layouts) ----
// A (16x32 f16): lane m holds K {0..7,16..23}; lane m+16 holds K {8..15,24..31}
__device__ __forceinline__ int a_swz(int block_r, int k, int m) {
  int bk = k >> 5, kin = k & 31;
  int lane = m + (((kin >> 3) & 1) << 4);
  int pos  = (kin & 7) + ((kin >> 4) << 3);
  return (((block_r * NB_K + bk) * 32 + lane) << 4) + pos;
}
// B (32x16 f16): lane n holds K {0..7,16..23}; lane n+16 holds K {8..15,24..31}
__device__ __forceinline__ int b_swz(int k, int n) {
  int bk = k >> 5, kin = k & 31;
  int bn = n >> 4, nin = n & 15;
  int lane = nin + (((kin >> 3) & 1) << 4);
  int pos  = (kin & 7) + ((kin >> 4) << 3);
  return (((bk * NB_N + bn) * 32 + lane) << 4) + pos;
}

// ---- CDNA5 async global->LDS copy (ASYNCcnt-tracked DMA) ----
__device__ __forceinline__ void async_copy_b128(unsigned lds_addr, const void* gaddr) {
  asm volatile("global_load_async_to_lds_b128 %0, %1, off"
               :: "v"(lds_addr), "v"((unsigned long long)(uintptr_t)gaddr)
               : "memory");
}
__device__ __forceinline__ void wait_asynccnt0() {
  asm volatile("s_wait_asynccnt 0x0" ::: "memory");
}

// =====================================================================
// Kernel 1: generate block-complex QFT matrix, f16 hi/lo, pre-swizzled.
// B = [[Re, Im], [-Im, Re]],  Re/Im[j][n] = cos/sin(2*pi*(j*n mod 1024)/1024)/32
// =====================================================================
__global__ __launch_bounds__(512) void qft_b_kernel(_Float16* __restrict__ Bhi,
                                                    _Float16* __restrict__ Blo) {
  const float w0 = 6.283185307179586f / 1024.f;
  int t0 = blockIdx.x * 512 + threadIdx.x;
  #pragma unroll
  for (int i = 0; i < 4; ++i) {
    int e = t0 + i * (2048 * 512);
    int k = e >> 11, n = e & 2047;
    int prod = ((k & 1023) * (n & 1023)) & 1023;   // exact integer phase
    float sv, cv;
    __sincosf(w0 * (float)prod, &sv, &cv);
    float re = cv * 0.03125f, im = sv * 0.03125f;
    float v = (k < DIM) ? ((n < DIM) ? re : im)
                        : ((n < DIM) ? -im : re);
    _Float16 hi = (_Float16)v;
    _Float16 lo = (_Float16)(v - (float)hi);
    int idx = b_swz(k, n);
    Bhi[idx] = hi;
    Blo[idx] = lo;
  }
}

// =====================================================================
// Kernel 2: full circuit per state row in LDS (normalize, RX,RY,RZ,CNOT,RY),
// then emit row as [re|im] (len 2048) in f16 hi/lo WMMA-A swizzled layout.
// =====================================================================
__global__ __launch_bounds__(256) void circuit_kernel(const float* __restrict__ x,
                                                      const float* __restrict__ w,
                                                      _Float16* __restrict__ Ahi,
                                                      _Float16* __restrict__ Alo) {
  __shared__ float sre[DIM];
  __shared__ float sim[DIM];
  __shared__ float red[8];
  const int tid = threadIdx.x;
  const int row = blockIdx.x;
  const float* xr = x + (size_t)row * DIM;

  // ---- normalize ----
  float xv[4]; float ss = 0.f;
  #pragma unroll
  for (int j = 0; j < 4; ++j) { xv[j] = xr[tid + (j << 8)]; ss += xv[j] * xv[j]; }
  #pragma unroll
  for (int off = 16; off; off >>= 1) ss += __shfl_xor(ss, off, 32);
  if ((tid & 31) == 0) red[tid >> 5] = ss;
  __syncthreads();
  float tot = 0.f;
  #pragma unroll
  for (int j = 0; j < 8; ++j) tot += red[j];
  const float scale = 1.f / fmaxf(sqrtf(tot), 1e-8f);
  #pragma unroll
  for (int j = 0; j < 4; ++j) { sre[tid + (j << 8)] = xv[j] * scale; sim[tid + (j << 8)] = 0.f; }
  __syncthreads();

  // ---- RX layer: [[c,-is],[-is,c]] ----
  for (int q = 0; q < NQ; ++q) {
    float th = 0.5f * w[q * 4 + 0];
    float c = __cosf(th), s = __sinf(th);
    int b = 9 - q, msk = 1 << b;
    #pragma unroll
    for (int r = 0; r < 2; ++r) {
      int p = tid + (r << 8);
      int i0 = ((p >> b) << (b + 1)) | (p & (msk - 1));
      int i1 = i0 | msk;
      float r0 = sre[i0], m0 = sim[i0], r1 = sre[i1], m1 = sim[i1];
      sre[i0] = c * r0 + s * m1;   sim[i0] = c * m0 - s * r1;
      sre[i1] = s * m0 + c * r1;   sim[i1] = -s * r0 + c * m1;
    }
    __syncthreads();
  }

  // ---- RY layers (widx 1 and later 3): [[c,-s],[s,c]] ----
  auto ry_layer = [&](int widx) {
    for (int q = 0; q < NQ; ++q) {
      float th = 0.5f * w[q * 4 + widx];
      float c = __cosf(th), s = __sinf(th);
      int b = 9 - q, msk = 1 << b;
      #pragma unroll
      for (int r = 0; r < 2; ++r) {
        int p = tid + (r << 8);
        int i0 = ((p >> b) << (b + 1)) | (p & (msk - 1));
        int i1 = i0 | msk;
        float r0 = sre[i0], m0 = sim[i0], r1 = sre[i1], m1 = sim[i1];
        sre[i0] = c * r0 - s * r1;   sim[i0] = c * m0 - s * m1;
        sre[i1] = s * r0 + c * r1;   sim[i1] = s * m0 + c * m1;
      }
      __syncthreads();
    }
  };
  ry_layer(1);

  // ---- RZ layer: diag(e^{-it/2}, e^{+it/2}); elementwise ----
  for (int q = 0; q < NQ; ++q) {
    float th = 0.5f * w[q * 4 + 2];
    float c = __cosf(th), s = __sinf(th);
    int b = 9 - q;
    #pragma unroll
    for (int j = 0; j < 4; ++j) {
      int i = tid + (j << 8);
      float sg = ((i >> b) & 1) ? s : -s;
      float r0 = sre[i], m0 = sim[i];
      sre[i] = r0 * c - m0 * sg;
      sim[i] = m0 * c + r0 * sg;
    }
  }
  __syncthreads();

  // ---- CNOT ring: control q (bit 9-q), target (q+1)%10 ----
  for (int q = 0; q < NQ; ++q) {
    int tq = (q + 1) % NQ;
    int bc = 9 - q, bt = 9 - tq;
    #pragma unroll
    for (int j = 0; j < 4; ++j) {
      int i = tid + (j << 8);
      if (((i >> bc) & 1) && !((i >> bt) & 1)) {
        int ip = i | (1 << bt);
        float tr = sre[i], tm = sim[i];
        sre[i] = sre[ip];  sim[i] = sim[ip];
        sre[ip] = tr;      sim[ip] = tm;
      }
    }
    __syncthreads();
  }

  ry_layer(3);

  // ---- emit [re|im] row, f16 hi/lo split, WMMA-A swizzled ----
  const int block_r = row >> 4, m = row & 15;
  #pragma unroll
  for (int j = 0; j < 8; ++j) {
    int k = tid + (j << 8);                       // 0..2047
    float v = (k < DIM) ? sre[k] : sim[k - DIM];
    _Float16 hi = (_Float16)v;
    _Float16 lo = (_Float16)(v - (float)hi);
    int idx = a_swz(block_r, k, m);
    Ahi[idx] = hi;
    Alo[idx] = lo;
  }
}

// =====================================================================
// Kernel 3: 32-row x 2048-col GEMM stripe via v_wmma_f32_16x16x32_f16
// (3-term hi/lo split). A tiles staged in LDS via async global->LDS DMA
// (double-buffered, issued by wave 0, drained with s_wait_asynccnt).
// Fused expectation epilogue -> out[32x30], two 16-row passes reusing
// the same 64 KB LDS tile (which also aliases the A staging buffer).
// 512 threads = 16 waves; wave w owns n-tiles [8w, 8w+8).
// =====================================================================
__global__ __launch_bounds__(512) void gemm_expect_kernel(const _Float16* __restrict__ Ahip,
                                                          const _Float16* __restrict__ Alop,
                                                          const _Float16* __restrict__ Bhip,
                                                          const _Float16* __restrict__ Blop,
                                                          float* __restrict__ out) {
  // 64 KB shared: during K-loop, first 8 KB = A double buffer (2 x 4 KB);
  // after K-loop, whole array = 16x1024 complex f16 epilogue tile.
  __shared__ __align__(32) char smem[65536];
  const int tid  = threadIdx.x;
  const int lane = tid & 31, wv = tid >> 5;
  const int block_r2 = blockIdx.x;                 // 32-row block
  const v16h* Bh = (const v16h*)Bhip;
  const v16h* Bl = (const v16h*)Blop;

  // stage A tiles for k-block kb into buf: layout [(rb*2+hl)*32 + lane] v16h
  auto stage_A = [&](char* buf, int kb) {
    #pragma unroll
    for (int rb = 0; rb < 2; ++rb) {
      #pragma unroll
      for (int hl = 0; hl < 2; ++hl) {
        const _Float16* src = (hl ? Alop : Ahip) +
            (size_t)(((block_r2 * 2 + rb) * NB_K + kb) * 32 + lane) * 16;
        unsigned dst = (unsigned)(uintptr_t)(buf + ((rb * 2 + hl) * 32 + lane) * 32);
        async_copy_b128(dst, src);
        async_copy_b128(dst + 16, (const char*)src + 16);
      }
    }
  };

  v8f acc[2][8];
  const v8f vzero = {0.f, 0.f, 0.f, 0.f, 0.f, 0.f, 0.f, 0.f};
  #pragma unroll
  for (int rb = 0; rb < 2; ++rb)
    #pragma unroll
    for (int t = 0; t < 8; ++t) acc[rb][t] = vzero;

  // prologue: wave 0 stages kb=0
  if (wv == 0) { stage_A(smem, 0); wait_asynccnt0(); }
  __syncthreads();

  for (int kb = 0; kb < NB_K; ++kb) {
    if (wv == 0 && kb + 1 < NB_K) stage_A(smem + ((kb + 1) & 1) * 4096, kb + 1);

    const char* cur = smem + (kb & 1) * 4096;
    v16h a_h[2], a_l[2];
    #pragma unroll
    for (int rb = 0; rb < 2; ++rb) {
      a_h[rb] = *(const v16h*)(cur + ((rb * 2 + 0) * 32 + lane) * 32);
      a_l[rb] = *(const v16h*)(cur + ((rb * 2 + 1) * 32 + lane) * 32);
    }
    #pragma unroll
    for (int t = 0; t < 8; ++t) {
      int nb = (wv << 3) + t;
      int bidx = (kb * NB_N + nb) * 32 + lane;
      v16h bh = Bh[bidx];
      v16h bl = Bl[bidx];
      #pragma unroll
      for (int rb = 0; rb < 2; ++rb) {
        acc[rb][t] = __builtin_amdgcn_wmma_f32_16x16x32_f16(false, a_h[rb], false, bh, (short)0, acc[rb][t], false, false);
        acc[rb][t] = __builtin_amdgcn_wmma_f32_16x16x32_f16(false, a_h[rb], false, bl, (short)0, acc[rb][t], false, false);
        acc[rb][t] = __builtin_amdgcn_wmma_f32_16x16x32_f16(false, a_l[rb], false, bh, (short)0, acc[rb][t], false, false);
      }
    }
    if (kb + 1 < NB_K) {
      if (wv == 0) wait_asynccnt0();
      __syncthreads();           // next A buffer ready; prior buffer free
    }
  }

  __syncthreads();               // K-loop LDS use done; reuse smem as tile
  _Float16 (*tile)[2048] = (_Float16 (*)[2048])smem;

  #pragma unroll 1
  for (int rb = 0; rb < 2; ++rb) {
    // ---- dump accumulators (cols 0..1023 = re, 1024.. = im interleaved) ----
    #pragma unroll
    for (int t = 0; t < 8; ++t) {
      int nb = (wv << 3) + t;
      int isim = (nb >= 64) ? 1 : 0;
      int n  = ((nb & 63) << 4) + (lane & 15);
      int mb = (lane >> 4) << 3;               // C layout: VGPR j -> M=j (+8 hi lanes)
      #pragma unroll
      for (int j = 0; j < 8; ++j)
        tile[mb + j][(n << 1) + isim] = (_Float16)acc[rb][t][j];
    }
    __syncthreads();

    // ---- expectations: wave wv handles row m = wv ----
    const int m = wv;
    const int grow = (block_r2 << 5) + (rb << 4) + m;
    for (int q = 0; q < NQ; ++q) {
      int b = 9 - q, msk = 1 << b;
      float cr = 0.f, ci = 0.f, ez = 0.f;
      for (int p = lane; p < 512; p += 32) {
        int n0 = ((p >> b) << (b + 1)) | (p & (msk - 1));
        int n1 = n0 | msk;
        float r0 = (float)tile[m][n0 << 1],  i0 = (float)tile[m][(n0 << 1) + 1];
        float r1 = (float)tile[m][n1 << 1],  i1 = (float)tile[m][(n1 << 1) + 1];
        cr += r0 * r1 + i0 * i1;               // Re(conj(s0)*s1)
        ci += r0 * i1 - i0 * r1;               // Im(conj(s0)*s1)
        ez += (r0 * r0 + i0 * i0) - (r1 * r1 + i1 * i1);
      }
      #pragma unroll
      for (int off = 16; off; off >>= 1) {
        cr += __shfl_xor(cr, off, 32);
        ci += __shfl_xor(ci, off, 32);
        ez += __shfl_xor(ez, off, 32);
      }
      if (lane == 0) {
        out[grow * 30 + q]      = 2.f * cr;
        out[grow * 30 + 10 + q] = 2.f * ci;
        out[grow * 30 + 20 + q] = ez;
      }
    }
    __syncthreads();             // tile free for next rb pass
  }
}

// =====================================================================
extern "C" void kernel_launch(void* const* d_in, const int* in_sizes, int n_in,
                              void* d_out, int out_size, void* d_ws, size_t ws_size,
                              hipStream_t stream) {
  (void)in_sizes; (void)n_in; (void)out_size; (void)ws_size;
  const float* x = (const float*)d_in[0];
  const float* w = (const float*)d_in[1];
  float* out = (float*)d_out;
  char* ws = (char*)d_ws;
  // workspace map (80 MB total): A hi/lo 32MB each, B hi/lo 8MB each
  _Float16* Ahi = (_Float16*)(ws);
  _Float16* Alo = (_Float16*)(ws + (size_t)32 * 1024 * 1024);
  _Float16* Bhi = (_Float16*)(ws + (size_t)64 * 1024 * 1024);
  _Float16* Blo = (_Float16*)(ws + (size_t)72 * 1024 * 1024);

  qft_b_kernel<<<2048, 512, 0, stream>>>(Bhi, Blo);
  circuit_kernel<<<BATCH, 256, 0, stream>>>(x, w, Ahi, Alo);
  gemm_expect_kernel<<<BATCH / 32, 512, 0, stream>>>(Ahi, Alo, Bhi, Blo, out);
}
